// ViMBlock_69870527972109
// MI455X (gfx1250) — matrix operand order repaired and checked by reading, not compiled, not run on hardware
//
#include <hip/hip_runtime.h>
#include <math.h>

// ---------------- problem constants (from reference) ----------------
#define BB 4
#define LL 4096
#define DD 1024
#define NN 64
#define KK 3
#define MM (BB * LL)          // 16384 rows
#define EPSF 1e-6f
// chunked scan decomposition
#define SCH 64                // chunks per sequence
#define SCL (LL / SCH)        // chunk length = 64

typedef __attribute__((ext_vector_type(16))) __bf16 bf16x16;
typedef __attribute__((ext_vector_type(8)))  float  f32x8;

// ======================================================================
// small helper kernels
// ======================================================================
__global__ void k_cvt_f32_bf16(const float* __restrict__ s,
                               __bf16* __restrict__ d, int n) {
    int i = blockIdx.x * blockDim.x + threadIdx.x;
    if (i < n) d[i] = (__bf16)s[i];
}

// woutcat[d*128 + k] = k<64 ? f_wout[d,k] : b_wout[d,k-64]   (d:0..1023)
__global__ void k_wout_concat(const float* __restrict__ fw,
                              const float* __restrict__ bw,
                              __bf16* __restrict__ dst) {
    int i = blockIdx.x * blockDim.x + threadIdx.x;       // 1024*128
    if (i >= DD * 2 * NN) return;
    int k = i & 127, d = i >> 7;
    float v = (k < NN) ? fw[d * NN + k] : bw[d * NN + (k - NN)];
    dst[i] = (__bf16)v;
}

// ======================================================================
// RMSNorm:  Tn = T / sqrt(mean(T^2)+eps) * scale   -> bf16
// one block (256 thr) per row of D=1024
// ======================================================================
__global__ void k_rmsnorm(const float* __restrict__ T,
                          const float* __restrict__ scale,
                          __bf16* __restrict__ Tn) {
    __shared__ float part[8];
    __shared__ float stot;
    const int m = blockIdx.x;
    const float* row = T + (size_t)m * DD;
    float s = 0.f;
    #pragma unroll
    for (int i = 0; i < 4; ++i) {
        float v = row[threadIdx.x + i * 256];
        s += v * v;
    }
    #pragma unroll
    for (int o = 16; o > 0; o >>= 1) s += __shfl_down(s, o, 32);
    int lane = threadIdx.x & 31, wv = threadIdx.x >> 5;
    if (lane == 0) part[wv] = s;
    __syncthreads();
    if (threadIdx.x == 0) {
        float t = 0.f;
        #pragma unroll
        for (int w = 0; w < 8; ++w) t += part[w];
        stot = rsqrtf(t * (1.0f / DD) + EPSF);
    }
    __syncthreads();
    float inv = stot;
    __bf16* orow = Tn + (size_t)m * DD;
    #pragma unroll
    for (int i = 0; i < 4; ++i) {
        int e = threadIdx.x + i * 256;
        orow[e] = (__bf16)(row[e] * inv * scale[e]);
    }
}

// ======================================================================
// depthwise conv1d, K=3, pad=1 over L (per channel), + bias -> bf16
// ======================================================================
__global__ void k_dwconv(const __bf16* __restrict__ x,
                         const float* __restrict__ w,   // (D,1,3)
                         const float* __restrict__ bias,
                         __bf16* __restrict__ y) {
    int idx = blockIdx.x * blockDim.x + threadIdx.x;     // B*L*D
    if (idx >= BB * LL * DD) return;
    int d = idx % DD;
    int l = (idx / DD) % LL;
    int b = idx / (DD * LL);
    float acc = bias[d];
    #pragma unroll
    for (int k = 0; k < KK; ++k) {
        int ll = l + k - 1;
        if (ll >= 0 && ll < LL)
            acc += (float)x[((size_t)(b * LL + ll)) * DD + d] * w[d * KK + k];
    }
    y[idx] = (__bf16)acc;
}

// ======================================================================
// bidirectional diagonal SSM scan (ZOH), chunked 3-pass formulation.
// u: (M,128) f32, cols 0..63 forward, 64..127 backward.
// out ys = h*Cp -> bf16 (M,128).
// Serial depth 3*64 instead of 4096; lane == n => coalesced 128B reads.
// ======================================================================
__device__ inline void ssm_params(int n, int dir,
                                  const float* fA, const float* fB,
                                  const float* fC, const float* fdt,
                                  const float* bA, const float* bB,
                                  const float* bC, const float* bdt,
                                  float& Abar, float& Bbar, float& Cp,
                                  float& Achk) {
    const float* logA = dir ? bA : fA;
    const float* Bp   = dir ? bB : fB;
    const float* Cv   = dir ? bC : fC;
    float dt   = expf((dir ? bdt : fdt)[0]);
    float A    = -expf(logA[n]);
    Abar = expf(A * dt);
    float frac = (fabsf(A) < EPSF) ? dt : (Abar - 1.0f) / A;
    Bbar = frac * Bp[n];
    Cp   = Cv[n];
    Achk = expf(A * dt * (float)SCL);   // Abar^SCL
}

// Pass A: local scan per chunk from h=0; emit chunk carry.
// threads: dir*b*chunk*n = 2*4*64*64 = 32768
__global__ void k_scan_chunk(const float* __restrict__ u,
                             float* __restrict__ carry,
                             const float* fA, const float* fB, const float* fC,
                             const float* fdt,
                             const float* bA, const float* bB, const float* bC,
                             const float* bdt) {
    int tid = blockIdx.x * blockDim.x + threadIdx.x;
    if (tid >= 2 * BB * SCH * NN) return;
    int n   = tid & (NN - 1);
    int b   = (tid >> 6) & (BB - 1);
    int dir = (tid >> 8) & 1;
    int c   = tid >> 9;

    float Abar, Bbar, Cp, Achk;
    ssm_params(n, dir, fA, fB, fC, fdt, bA, bB, bC, bdt, Abar, Bbar, Cp, Achk);

    int col = n + (dir ? NN : 0);
    float h = 0.f;
    for (int i = 0; i < SCL; ++i) {
        int l = dir ? (LL - 1 - (c * SCL + i)) : (c * SCL + i);
        size_t idx = ((size_t)(b * LL + l)) * 128 + col;
        h = h * Abar + u[idx] * Bbar;
    }
    carry[(((size_t)dir * BB + b) * SCH + c) * NN + n] = h;
}

// Pass B: serial prefix over 64 chunk carries per (dir,b,n); 512 threads.
// hpre[c] = true state entering chunk c.
__global__ void k_scan_combine(const float* __restrict__ carry,
                               float* __restrict__ hpre,
                               const float* fA, const float* fB, const float* fC,
                               const float* fdt,
                               const float* bA, const float* bB, const float* bC,
                               const float* bdt) {
    int tid = blockIdx.x * blockDim.x + threadIdx.x;
    if (tid >= 2 * BB * NN) return;
    int n   = tid & (NN - 1);
    int b   = (tid >> 6) & (BB - 1);
    int dir = tid >> 8;

    float Abar, Bbar, Cp, Achk;
    ssm_params(n, dir, fA, fB, fC, fdt, bA, bB, bC, bdt, Abar, Bbar, Cp, Achk);

    float H = 0.f;
    for (int c = 0; c < SCH; ++c) {
        size_t idx = (((size_t)dir * BB + b) * SCH + c) * NN + n;
        hpre[idx] = H;
        H = H * Achk + carry[idx];
    }
}

// Pass C: rerun local recurrence seeded with true incoming state; emit ys.
__global__ void k_scan_apply(const float* __restrict__ u,
                             const float* __restrict__ hpre,
                             __bf16* __restrict__ ys,
                             const float* fA, const float* fB, const float* fC,
                             const float* fdt,
                             const float* bA, const float* bB, const float* bC,
                             const float* bdt) {
    int tid = blockIdx.x * blockDim.x + threadIdx.x;
    if (tid >= 2 * BB * SCH * NN) return;
    int n   = tid & (NN - 1);
    int b   = (tid >> 6) & (BB - 1);
    int dir = (tid >> 8) & 1;
    int c   = tid >> 9;

    float Abar, Bbar, Cp, Achk;
    ssm_params(n, dir, fA, fB, fC, fdt, bA, bB, bC, bdt, Abar, Bbar, Cp, Achk);

    int col = n + (dir ? NN : 0);
    float h = hpre[(((size_t)dir * BB + b) * SCH + c) * NN + n];
    for (int i = 0; i < SCL; ++i) {
        int l = dir ? (LL - 1 - (c * SCL + i)) : (c * SCL + i);
        size_t idx = ((size_t)(b * LL + l)) * 128 + col;
        h = h * Abar + u[idx] * Bbar;
        ys[idx] = (__bf16)(h * Cp);
    }
}

// ======================================================================
// WMMA GEMM:  C(MxN) = A(MxK,bf16,row-major) * Bw(NxK,bf16,row-major)^T
// block = 256 thr = 8 waves; wave tile 32x32 (2x2 wmma); block tile 128x64
// grid = (N/64, M/128)
// Epilogues:
//  0: split+gate   v=acc+bias[j]; j<DD -> outb=bf16(v); else outf=sigmoid(v)
//  1: bias->bf16   outb = bf16(acc+bias[j])
//  2: plain f32    outf = acc
//  3: gate mult    outb = bf16(acc * gate[m*DD+j])
//  4: residual     outf = resid[m*N+j] + acc + bias[j]
// ======================================================================
__device__ inline bf16x16 load_afrag(const __bf16* A, int lda, int row,
                                     int lane, int k0) {
    int g = lane >> 4, m = lane & 15;
    const __bf16* p = A + (size_t)(row + m) * lda + k0 + g * 16;
    bf16x16 a;
    #pragma unroll
    for (int r = 0; r < 8; ++r) {
        int kp = r << 1;
        a[2 * r]     = p[kp];
        a[2 * r + 1] = p[kp + 1];
    }
    return a;
}

__device__ inline bf16x16 load_bfrag(const __bf16* Bw, int ldb, int col,
                                     int lane, int k0) {
    int g = lane >> 4, n = lane & 15;
    const __bf16* p = Bw + (size_t)(col + n) * ldb + k0 + g * 16;
    return *(const bf16x16*)p;   // 32B aligned: k0%32==0, rows 16-elt aligned
}

template <int EPI>
__global__ void k_gemm(const __bf16* __restrict__ A,
                       const __bf16* __restrict__ Bw,
                       int M, int N, int K,
                       const float* __restrict__ bias,
                       const float* __restrict__ gate,
                       const float* __restrict__ resid,
                       __bf16* __restrict__ outb,
                       float* __restrict__ outf) {
    const int lane   = threadIdx.x & 31;
    const int waveId = threadIdx.x >> 5;          // 0..7
    const int wm = waveId & 3, wn = waveId >> 2;  // 4 x 2 wave grid
    const int mBase = blockIdx.y * 128 + wm * 32;
    const int nBase = blockIdx.x * 64 + wn * 32;

    f32x8 c00 = {}, c01 = {}, c10 = {}, c11 = {};

    for (int k0 = 0; k0 < K; k0 += 32) {
        bf16x16 a0 = load_afrag(A, K, mBase,      lane, k0);
        bf16x16 a1 = load_afrag(A, K, mBase + 16, lane, k0);
        bf16x16 b0 = load_bfrag(Bw, K, nBase,      lane, k0);
        bf16x16 b1 = load_bfrag(Bw, K, nBase + 16, lane, k0);
        c00 = __builtin_amdgcn_wmma_f32_16x16x32_bf16(false, a0, false, b0,
                                                      (short)0, c00, false, false);
        c01 = __builtin_amdgcn_wmma_f32_16x16x32_bf16(false, a0, false, b1,
                                                      (short)0, c01, false, false);
        c10 = __builtin_amdgcn_wmma_f32_16x16x32_bf16(false, a1, false, b0,
                                                      (short)0, c10, false, false);
        c11 = __builtin_amdgcn_wmma_f32_16x16x32_bf16(false, a1, false, b1,
                                                      (short)0, c11, false, false);
    }

    const int g = lane >> 4, nl = lane & 15;
    #pragma unroll
    for (int mi = 0; mi < 2; ++mi) {
        #pragma unroll
        for (int ni = 0; ni < 2; ++ni) {
            f32x8 c = (mi == 0) ? (ni == 0 ? c00 : c01)
                                : (ni == 0 ? c10 : c11);
            #pragma unroll
            for (int r = 0; r < 8; ++r) {
                int m = mBase + mi * 16 + g * 8 + r;
                int j = nBase + ni * 16 + nl;
                float acc = c[r];
                if constexpr (EPI == 0) {
                    float v = acc + bias[j];
                    if (j < DD)
                        outb[(size_t)m * DD + j] = (__bf16)v;
                    else
                        outf[(size_t)m * DD + (j - DD)] =
                            1.0f / (1.0f + expf(-v));
                } else if constexpr (EPI == 1) {
                    outb[(size_t)m * N + j] = (__bf16)(acc + bias[j]);
                } else if constexpr (EPI == 2) {
                    outf[(size_t)m * N + j] = acc;
                } else if constexpr (EPI == 3) {
                    outb[(size_t)m * DD + j] =
                        (__bf16)(acc * gate[(size_t)m * DD + j]);
                } else {  // EPI == 4
                    outf[(size_t)m * N + j] =
                        resid[(size_t)m * N + j] + acc + bias[j];
                }
            }
        }
    }
}

// ======================================================================
// launch
// ======================================================================
extern "C" void kernel_launch(void* const* d_in, const int* in_sizes, int n_in,
                              void* d_out, int out_size, void* d_ws, size_t ws_size,
                              hipStream_t stream) {
    (void)in_sizes; (void)n_in; (void)out_size; (void)ws_size;
    const float* T      = (const float*)d_in[0];
    const float* nscale = (const float*)d_in[1];
    const float* in_w   = (const float*)d_in[2];
    const float* in_b   = (const float*)d_in[3];
    const float* dw_w   = (const float*)d_in[4];
    const float* dw_b   = (const float*)d_in[5];
    const float* pw_w   = (const float*)d_in[6];
    const float* pw_b   = (const float*)d_in[7];
    const float* f_win  = (const float*)d_in[8];
    const float* f_wout = (const float*)d_in[9];
    const float* f_logA = (const float*)d_in[10];
    const float* f_B    = (const float*)d_in[11];
    const float* f_C    = (const float*)d_in[12];
    const float* f_dt   = (const float*)d_in[13];
    const float* b_win  = (const float*)d_in[14];
    const float* b_wout = (const float*)d_in[15];
    const float* b_logA = (const float*)d_in[16];
    const float* b_B    = (const float*)d_in[17];
    const float* b_C    = (const float*)d_in[18];
    const float* b_dt   = (const float*)d_in[19];
    const float* out_w  = (const float*)d_in[20];
    const float* out_b  = (const float*)d_in[21];
    float* out = (float*)d_out;

    // ---- workspace layout (aliased) ----
    char* ws = (char*)d_ws;
    size_t off = 0;
    auto alloc = [&](size_t bytes) {
        void* p = ws + off;
        off += (bytes + 255) & ~(size_t)255;
        return p;
    };
    __bf16* buf0    = (__bf16*)alloc((size_t)MM * DD * 2);   // Tn, later xdw
    __bf16* buf1    = (__bf16*)alloc((size_t)MM * DD * 2);   // x,  later ygate
    __bf16* xconv   = (__bf16*)alloc((size_t)MM * DD * 2);
    float*  gate    = (float*) alloc((size_t)MM * DD * 4);
    __bf16* inw_bf  = (__bf16*)alloc((size_t)2 * DD * DD * 2);
    __bf16* pww_bf  = (__bf16*)alloc((size_t)DD * DD * 2);
    __bf16* outw_bf = (__bf16*)alloc((size_t)DD * DD * 2);
    __bf16* wincat  = (__bf16*)alloc((size_t)2 * NN * DD * 2);
    __bf16* woutcat = (__bf16*)alloc((size_t)DD * 2 * NN * 2);
    float*  u       = (float*) alloc((size_t)MM * 2 * NN * 4);
    __bf16* ysc     = (__bf16*)alloc((size_t)MM * 2 * NN * 2);
    float*  carry   = (float*) alloc((size_t)2 * BB * SCH * NN * 4);
    float*  hpre    = (float*) alloc((size_t)2 * BB * SCH * NN * 4);

    __bf16* Tn    = buf0;
    __bf16* xbf   = buf1;
    __bf16* xdw   = buf0;   // Tn dead after G1
    __bf16* ygate = buf1;   // x  dead after conv

    const dim3 blk(256);

    // 1) weight conversion
    k_cvt_f32_bf16<<<(2 * DD * DD + 255) / 256, blk, 0, stream>>>(in_w, inw_bf, 2 * DD * DD);
    k_cvt_f32_bf16<<<(DD * DD + 255) / 256, blk, 0, stream>>>(pw_w, pww_bf, DD * DD);
    k_cvt_f32_bf16<<<(DD * DD + 255) / 256, blk, 0, stream>>>(out_w, outw_bf, DD * DD);
    k_cvt_f32_bf16<<<(NN * DD + 255) / 256, blk, 0, stream>>>(f_win, wincat, NN * DD);
    k_cvt_f32_bf16<<<(NN * DD + 255) / 256, blk, 0, stream>>>(b_win, wincat + NN * DD, NN * DD);
    k_wout_concat<<<(DD * 2 * NN + 255) / 256, blk, 0, stream>>>(f_wout, b_wout, woutcat);

    // 2) RMSNorm -> Tn (bf16)
    k_rmsnorm<<<MM, blk, 0, stream>>>(T, nscale, Tn);

    // 3) G1: xz = Tn @ in_w^T ; split into x (bf16) and gate = sigmoid (f32)
    k_gemm<0><<<dim3(2 * DD / 64, MM / 128), blk, 0, stream>>>(
        Tn, inw_bf, MM, 2 * DD, DD, in_b, nullptr, nullptr, xbf, gate);

    // 4) depthwise conv
    k_dwconv<<<(BB * LL * DD + 255) / 256, blk, 0, stream>>>(xbf, dw_w, dw_b, xdw);

    // 5) G2: pointwise conv + bias -> xconv (bf16)
    k_gemm<1><<<dim3(DD / 64, MM / 128), blk, 0, stream>>>(
        xdw, pww_bf, MM, DD, DD, pw_b, nullptr, nullptr, xconv, nullptr);

    // 6) G3: u = xconv @ [f_win;b_win]^T  (M x 128, f32)
    k_gemm<2><<<dim3(2 * NN / 64, MM / 128), blk, 0, stream>>>(
        xconv, wincat, MM, 2 * NN, DD, nullptr, nullptr, nullptr, nullptr, u);

    // 7) bidirectional scan, 3-pass chunked -> ys (bf16, already * Cp)
    k_scan_chunk<<<(2 * BB * SCH * NN + 255) / 256, blk, 0, stream>>>(
        u, carry, f_logA, f_B, f_C, f_dt, b_logA, b_B, b_C, b_dt);
    k_scan_combine<<<2, blk, 0, stream>>>(
        carry, hpre, f_logA, f_B, f_C, f_dt, b_logA, b_B, b_C, b_dt);
    k_scan_apply<<<(2 * BB * SCH * NN + 255) / 256, blk, 0, stream>>>(
        u, hpre, ysc, f_logA, f_B, f_C, f_dt, b_logA, b_B, b_C, b_dt);

    // 8) G4: y = (ys @ wout^T) * gate -> ygate (bf16)
    k_gemm<3><<<dim3(DD / 64, MM / 128), blk, 0, stream>>>(
        ysc, woutcat, MM, DD, 2 * NN, nullptr, gate, nullptr, ygate, nullptr);

    // 9) G5: out = T + ygate @ out_w^T + out_b
    k_gemm<4><<<dim3(DD / 64, MM / 128), blk, 0, stream>>>(
        ygate, outw_bf, MM, DD, DD, out_b, nullptr, T, nullptr, out);
}
